// CronRootMultiheadAttention_50929722196094
// MI455X (gfx1250) — matrix operand order repaired and checked by reading, not compiled, not run on hardware
//
#include <hip/hip_runtime.h>

// ---------------------------------------------------------------------------
// CronRootMultiheadAttention for MI455X (gfx1250, wave32, WMMA bf16, TDM)
//   B=4, S=4096, D=1024, H=16, hd=64, block b=64, nb=64
// Pipeline:
//   0) f32 -> bf16 converts (x, in_proj_w, out_proj_w)
//   1) QKV GEMM (bf16 WMMA, f32 acc, TDM double-buffered LDS tiles),
//      epilogue scatters q/k/v [B,H,S,64] bf16
//   2) attention: 1 workgroup per (b,h,block): WMMA scores (local + summary),
//      masked softmax (wave32 shfl reductions), WMMA @ V / @ summary-V
//   3) out-proj GEMM (bf16 WMMA, TDM), writes f32 result
// Workspace use: ~176 MB.
// ---------------------------------------------------------------------------

typedef __attribute__((ext_vector_type(16))) __bf16 v16bf;
typedef __attribute__((ext_vector_type(8)))  __bf16 v8bf;
typedef __attribute__((ext_vector_type(8)))  float  v8f;
typedef __attribute__((ext_vector_type(4)))  unsigned int u32x4;
typedef __attribute__((ext_vector_type(4)))  int    i32x4;
typedef __attribute__((ext_vector_type(8)))  int    i32x8;

#define WMMA_BF16(a, b, c) \
  __builtin_amdgcn_wmma_f32_16x16x32_bf16(false, (a), false, (b), (short)0, (c), false, false)

#define NEGINF (-1e30f)

#if defined(__gfx1250__) && __has_builtin(__builtin_amdgcn_tensor_load_to_lds)
#define USE_TDM 1
#else
#define USE_TDM 0
#endif

// A-fragment (16x32 bf16, row-major source): lane L -> row (L&15),
// K chunks at k0 + hi*8 (16B) and k0 + 16 + hi*8 (16B).  (ISA 7.12.2)
__device__ __forceinline__ v16bf load_fragA(const __bf16* base, int ld, int row0, int k0, int lane) {
  const __bf16* p = base + (row0 + (lane & 15)) * ld + k0 + ((lane >> 4) << 3);
  v8bf lo = *(const v8bf*)(p);
  v8bf hi = *(const v8bf*)(p + 16);
  v16bf f;
#pragma unroll
  for (int i = 0; i < 8; ++i) { f[i] = lo[i]; f[i + 8] = hi[i]; }
  return f;
}

// B-fragment (32x16 bf16). Source holds B^T row-major (row = N column of B).
// lane L -> column (L&15), K chunks at k0 + hi*8 (16B) and k0 + 16 + hi*8 (16B).
__device__ __forceinline__ v16bf load_fragB(const __bf16* base, int ld, int col0, int k0, int lane) {
  const __bf16* p = base + (col0 + (lane & 15)) * ld + k0 + ((lane >> 4) << 3);
  v8bf lo = *(const v8bf*)(p);
  v8bf hi = *(const v8bf*)(p + 16);
  v16bf f;
#pragma unroll
  for (int i = 0; i < 8; ++i) { f[i] = lo[i]; f[i + 8] = hi[i]; }
  return f;
}

#if USE_TDM
// TDM load of a 2D bf16 tile: 64 cols x 128 rows, row stride K elements,
// into LDS with 16B padding after each 128B row (=> 72-element LDS stride).
// D# layout per CDNA5 ISA.
__device__ __forceinline__ void tdm_load_tile(const __bf16* gptr, unsigned lds_base, int K) {
  unsigned long long ga = (unsigned long long)(const void*)gptr;
  u32x4 g0;
  g0[0] = 1u;                               // count=1, user descriptor
  g0[1] = lds_base;                         // lds_addr (bytes)
  g0[2] = (unsigned)ga;                     // global_addr[31:0]
  g0[3] = ((unsigned)(ga >> 32) & 0x01FFFFFFu) | 0x80000000u;  // addr[56:32] | type=2
  const unsigned TD1 = 1u << 20;            // generous tensor_dim1 (no OOB hit)
  i32x8 g1;
  g1[0] = (int)((1u << 16)                  // data_size = 2 bytes
              | (1u << 20)                  // pad_enable
              | (4u << 22)                  // pad_interval: 32 DWORDs (128 B)
              | (3u << 25));                // pad_amount: 4 DWORDs (16 B)
  g1[1] = (int)(((unsigned)K & 0xffffu) << 16);            // tensor_dim0[15:0]
  g1[2] = (int)(((unsigned)K >> 16) | ((TD1 & 0xffffu) << 16));
  g1[3] = (int)((TD1 >> 16) | (64u << 16)); // tensor_dim1[31:16] | tile_dim0=64
  g1[4] = 128;                              // tile_dim1 = 128 rows (tile_dim2=0)
  g1[5] = K;                                // tensor_dim0_stride[31:0]
  g1[6] = 0;
  g1[7] = 0;
  i32x4 z4 = {};
#if __clang_major__ >= 23
  i32x8 z8 = {};
  __builtin_amdgcn_tensor_load_to_lds(g0, g1, z4, z4, z8, 0);
#else
  __builtin_amdgcn_tensor_load_to_lds(g0, g1, z4, z4, 0);
#endif
}

__device__ __forceinline__ unsigned lds_addr_of(const void* p) {
  // Generic LDS pointer: addr[31:0] is the LDS byte offset (ISA §10.2).
  return (unsigned)(unsigned long long)p;
}
#endif

// ---------------------------------------------------------------------------
__global__ __launch_bounds__(256) void cvt_f32_bf16_kernel(const float* __restrict__ in,
                                                           __bf16* __restrict__ out, int n) {
  int i = blockIdx.x * 256 + threadIdx.x;
  if (i < n) out[i] = (__bf16)in[i];
}

// ---------------------------------------------------------------------------
// GEMM: C[M,N] = A[M,K](bf16) @ W[N,K]^T(bf16) + bias(f32)
// mode 0: scatter into q/k/v bf16 [B=4,H=16,S=4096,hd=64]   (N = 3072)
// mode 1: write f32 row-major [M,N]                          (N = 1024)
#define GT_LD 72
__global__ __launch_bounds__(256) void gemm_bf16_kernel(
    const __bf16* __restrict__ A, const __bf16* __restrict__ W,
    const float* __restrict__ bias, int K, int N, int mode,
    __bf16* __restrict__ qout, __bf16* __restrict__ kout, __bf16* __restrict__ vout,
    float* __restrict__ fout) {
  __shared__ __align__(16) __bf16 As[2][128 * GT_LD];
  __shared__ __align__(16) __bf16 Bs[2][128 * GT_LD];

  const int tid  = threadIdx.x;
  const int lane = tid & 31;
  const int wave = tid >> 5;          // 8 waves
  const int wm   = wave >> 1;         // 0..3 : rows wm*32
  const int wn   = wave & 1;          // 0..1 : cols wn*64
  const int m0   = blockIdx.y * 128;
  const int n0   = blockIdx.x * 128;

  v8f acc[2][4];
#pragma unroll
  for (int i = 0; i < 2; ++i)
#pragma unroll
    for (int j = 0; j < 4; ++j) acc[i][j] = (v8f)0.0f;

#if USE_TDM
  // --- TDM double-buffered pipeline: DMA tile k+1 while WMMA consumes tile k.
  if (tid < 32) {
    tdm_load_tile(&A[(size_t)m0 * K], lds_addr_of(&As[0][0]), K);
    tdm_load_tile(&W[(size_t)n0 * K], lds_addr_of(&Bs[0][0]), K);
  }
  int buf = 0;
  for (int k0 = 0; k0 < K; k0 += 64) {
    if (tid < 32) __builtin_amdgcn_s_wait_tensorcnt(0);
    __syncthreads();
    if (tid < 32 && (k0 + 64) < K) {
      tdm_load_tile(&A[(size_t)m0 * K + k0 + 64], lds_addr_of(&As[buf ^ 1][0]), K);
      tdm_load_tile(&W[(size_t)n0 * K + k0 + 64], lds_addr_of(&Bs[buf ^ 1][0]), K);
    }
    const __bf16* Ac = As[buf];
    const __bf16* Bc = Bs[buf];
#pragma unroll
    for (int kk = 0; kk < 2; ++kk) {
      v16bf a0 = load_fragA(Ac, GT_LD, wm * 32,      kk * 32, lane);
      v16bf a1 = load_fragA(Ac, GT_LD, wm * 32 + 16, kk * 32, lane);
#pragma unroll
      for (int j = 0; j < 4; ++j) {
        v16bf bf = load_fragB(Bc, GT_LD, wn * 64 + j * 16, kk * 32, lane);
        acc[0][j] = WMMA_BF16(a0, bf, acc[0][j]);
        acc[1][j] = WMMA_BF16(a1, bf, acc[1][j]);
      }
    }
    __syncthreads();
    buf ^= 1;
  }
#else
  // --- fallback: single-buffer VGPR-staged tiles
  for (int k0 = 0; k0 < K; k0 += 64) {
#pragma unroll
    for (int c = 0; c < 4; ++c) {
      int idx = tid + c * 256;        // 0..1023 uint4 slots
      int row = idx >> 3;
      int col = idx & 7;
      *(uint4*)&As[0][row * GT_LD + col * 8] = *(const uint4*)&A[(size_t)(m0 + row) * K + k0 + col * 8];
      *(uint4*)&Bs[0][row * GT_LD + col * 8] = *(const uint4*)&W[(size_t)(n0 + row) * K + k0 + col * 8];
    }
    __syncthreads();
#pragma unroll
    for (int kk = 0; kk < 2; ++kk) {
      v16bf a0 = load_fragA(As[0], GT_LD, wm * 32,      kk * 32, lane);
      v16bf a1 = load_fragA(As[0], GT_LD, wm * 32 + 16, kk * 32, lane);
#pragma unroll
      for (int j = 0; j < 4; ++j) {
        v16bf bf = load_fragB(Bs[0], GT_LD, wn * 64 + j * 16, kk * 32, lane);
        acc[0][j] = WMMA_BF16(a0, bf, acc[0][j]);
        acc[1][j] = WMMA_BF16(a1, bf, acc[1][j]);
      }
    }
    __syncthreads();
  }
#endif

  const int coln = lane & 15;
  const int hi8  = (lane >> 4) << 3;
#pragma unroll
  for (int i = 0; i < 2; ++i)
#pragma unroll
    for (int j = 0; j < 4; ++j) {
      int nn = n0 + wn * 64 + j * 16 + coln;
      float bv = bias[nn];
#pragma unroll
      for (int r = 0; r < 8; ++r) {
        int m = m0 + wm * 32 + i * 16 + r + hi8;
        float val = acc[i][j][r] + bv;
        if (mode == 0) {
          int which = nn >> 10;             // 0:q 1:k 2:v
          int h     = (nn & 1023) >> 6;
          int d     = nn & 63;
          int bbq   = m >> 12;
          int s     = m & 4095;
          size_t off = (((size_t)(bbq * 16 + h)) * 4096 + s) * 64 + d;
          __bf16 o = (__bf16)val;
          if (which == 0) qout[off] = o;
          else if (which == 1) kout[off] = o;
          else vout[off] = o;
        } else {
          fout[(size_t)m * N + nn] = val;
        }
      }
    }
}

// ---------------------------------------------------------------------------
// Attention: one workgroup (128 threads = 4 waves) per (b, h, block n).
#define A_LD 72
__global__ __launch_bounds__(128) void cron_attn_kernel(
    const __bf16* __restrict__ q, const __bf16* __restrict__ k,
    const __bf16* __restrict__ v, __bf16* __restrict__ o /* [B,S,D] */) {
  __shared__ __align__(16) __bf16 Qs[64 * A_LD];   // Q block (later: W_loc)
  __shared__ __align__(16) __bf16 Ks[64 * A_LD];   // K block (later: W_str)
  __shared__ __align__(16) __bf16 Ss[64 * A_LD];   // summary keys  (nb x hd)
  __shared__ __align__(16) __bf16 Vt[64 * A_LD];   // V^T  : Vt[d][kk]
  __shared__ __align__(16) __bf16 St[64 * A_LD];   // Vs^T : St[d][m]

  const int blk = blockIdx.x;           // 0 .. 4*16*64-1
  const int n   = blk & 63;             // block index in sequence
  const int h   = (blk >> 6) & 15;
  const int bb  = blk >> 10;

  const int tid  = threadIdx.x;
  const int lane = tid & 31;
  const int wave = tid >> 5;            // 0..3, rows m0 = wave*16

  const size_t head = ((size_t)(bb * 16 + h)) * 4096 * 64;
  const __bf16* qg = q + head + (size_t)n * 64 * 64;
  const __bf16* kg = k + head + (size_t)n * 64 * 64;
  const __bf16* vg = v + head + (size_t)n * 64 * 64;
  const __bf16* ksum = k + head + 63 * 64;   // summary row m at stride 4096
  const __bf16* vsum = v + head + 63 * 64;

#pragma unroll
  for (int c = 0; c < 4; ++c) {
    int idx = tid + c * 128;            // 0..511 uint4 slots per 64x64 tile
    int row = idx >> 3;
    int col = idx & 7;
    *(uint4*)&Qs[row * A_LD + col * 8] = *(const uint4*)&qg[row * 64 + col * 8];
    *(uint4*)&Ks[row * A_LD + col * 8] = *(const uint4*)&kg[row * 64 + col * 8];
    *(uint4*)&Ss[row * A_LD + col * 8] = *(const uint4*)&ksum[(size_t)row * 4096 + col * 8];
    uint4 vv = *(const uint4*)&vg[row * 64 + col * 8];
    uint4 sv = *(const uint4*)&vsum[(size_t)row * 4096 + col * 8];
    const __bf16* pv = (const __bf16*)&vv;
    const __bf16* ps = (const __bf16*)&sv;
#pragma unroll
    for (int j = 0; j < 8; ++j) {       // transposed stores
      Vt[(col * 8 + j) * A_LD + row] = pv[j];
      St[(col * 8 + j) * A_LD + row] = ps[j];
    }
  }
  __syncthreads();

  const int m0 = wave * 16;
  v16bf af[2];
  af[0] = load_fragA(Qs, A_LD, m0, 0, lane);
  af[1] = load_fragA(Qs, A_LD, m0, 32, lane);

  v8f cl[4], cs[4];
#pragma unroll
  for (int t = 0; t < 4; ++t) { cl[t] = (v8f)0.0f; cs[t] = (v8f)0.0f; }
#pragma unroll
  for (int t = 0; t < 4; ++t)
#pragma unroll
    for (int kk = 0; kk < 2; ++kk) {
      v16bf bk = load_fragB(Ks, A_LD, t * 16, kk * 32, lane);
      cl[t] = WMMA_BF16(af[kk], bk, cl[t]);
      v16bf bs = load_fragB(Ss, A_LD, t * 16, kk * 32, lane);
      cs[t] = WMMA_BF16(af[kk], bs, cs[t]);
    }

  // masks + scale (1/sqrt(64))
  const int coln = lane & 15;
  const int hi8  = (lane >> 4) << 3;
  const float scale = 0.125f;
#pragma unroll
  for (int t = 0; t < 4; ++t) {
    int colg = t * 16 + coln;
    bool strOK = (colg < n);            // summary block strictly previous
#pragma unroll
    for (int r = 0; r < 8; ++r) {
      int rowg = m0 + r + hi8;
      cl[t][r] = (rowg >= colg) ? cl[t][r] * scale : NEGINF;
      cs[t][r] = strOK ? cs[t][r] * scale : NEGINF;
    }
  }

  // row-wise softmax over 128 cols; rows live per (r, lane-half), cols in lanes
#pragma unroll
  for (int r = 0; r < 8; ++r) {
    float mx = NEGINF;
#pragma unroll
    for (int t = 0; t < 4; ++t) { mx = fmaxf(mx, cl[t][r]); mx = fmaxf(mx, cs[t][r]); }
#pragma unroll
    for (int s = 1; s < 16; s <<= 1) mx = fmaxf(mx, __shfl_xor(mx, s, 32));
    float sum = 0.0f;
#pragma unroll
    for (int t = 0; t < 4; ++t) {
      float e0 = __expf(cl[t][r] - mx); cl[t][r] = e0; sum += e0;
      float e1 = __expf(cs[t][r] - mx); cs[t][r] = e1; sum += e1;
    }
#pragma unroll
    for (int s = 1; s < 16; s <<= 1) sum += __shfl_xor(sum, s, 32);
    float inv = 1.0f / sum;
#pragma unroll
    for (int t = 0; t < 4; ++t) { cl[t][r] *= inv; cs[t][r] *= inv; }
  }

  __syncthreads();                      // done reading Qs/Ks everywhere
#pragma unroll
  for (int t = 0; t < 4; ++t)
#pragma unroll
    for (int r = 0; r < 8; ++r) {
      int rowg = m0 + r + hi8;
      Qs[rowg * A_LD + t * 16 + coln] = (__bf16)cl[t][r];  // W_loc
      Ks[rowg * A_LD + t * 16 + coln] = (__bf16)cs[t][r];  // W_str
    }
  __syncthreads();

  v16bf wl[2], ws[2];
  wl[0] = load_fragA(Qs, A_LD, m0, 0, lane);
  wl[1] = load_fragA(Qs, A_LD, m0, 32, lane);
  ws[0] = load_fragA(Ks, A_LD, m0, 0, lane);
  ws[1] = load_fragA(Ks, A_LD, m0, 32, lane);

  const size_t obase = (((size_t)bb * 4096) + (size_t)n * 64) * 1024 + h * 64;
#pragma unroll
  for (int t = 0; t < 4; ++t) {
    v8f acc = (v8f)0.0f;
#pragma unroll
    for (int kk = 0; kk < 2; ++kk) {
      v16bf bv = load_fragB(Vt, A_LD, t * 16, kk * 32, lane);
      acc = WMMA_BF16(wl[kk], bv, acc);
      v16bf b2 = load_fragB(St, A_LD, t * 16, kk * 32, lane);
      acc = WMMA_BF16(ws[kk], b2, acc);
    }
#pragma unroll
    for (int r = 0; r < 8; ++r) {
      int rowg = m0 + r + hi8;
      o[obase + (size_t)rowg * 1024 + t * 16 + coln] = (__bf16)acc[r];
    }
  }
}

// ---------------------------------------------------------------------------
extern "C" void kernel_launch(void* const* d_in, const int* in_sizes, int n_in,
                              void* d_out, int out_size, void* d_ws, size_t ws_size,
                              hipStream_t stream) {
  const float* x     = (const float*)d_in[0];   // [4,4096,1024]
  const float* w_in  = (const float*)d_in[1];   // [3072,1024]
  const float* b_in  = (const float*)d_in[2];   // [3072]
  const float* w_out = (const float*)d_in[3];   // [1024,1024]
  const float* b_out = (const float*)d_in[4];   // [1024]
  float* out = (float*)d_out;                   // [4,4096,1024] f32

  const int M = 4 * 4096;                       // 16384
  const size_t nX  = (size_t)M * 1024;          // 16,777,216
  const size_t nWi = (size_t)3072 * 1024;
  const size_t nWo = (size_t)1024 * 1024;

  char* p = (char*)d_ws;
  __bf16* xb   = (__bf16*)p; p += nX  * 2;
  __bf16* wib  = (__bf16*)p; p += nWi * 2;
  __bf16* wob  = (__bf16*)p; p += nWo * 2;
  __bf16* qb   = (__bf16*)p; p += nX  * 2;      // [4,16,4096,64]
  __bf16* kb   = (__bf16*)p; p += nX  * 2;
  __bf16* vb   = (__bf16*)p; p += nX  * 2;
  __bf16* attn = (__bf16*)p; p += nX  * 2;      // [4,4096,1024]

  // 0) converts
  cvt_f32_bf16_kernel<<<(unsigned)((nX  + 255) / 256), 256, 0, stream>>>(x,     xb,  (int)nX);
  cvt_f32_bf16_kernel<<<(unsigned)((nWi + 255) / 256), 256, 0, stream>>>(w_in,  wib, (int)nWi);
  cvt_f32_bf16_kernel<<<(unsigned)((nWo + 255) / 256), 256, 0, stream>>>(w_out, wob, (int)nWo);

  // 1) QKV GEMM: [16384,1024] x [3072,1024]^T -> scatter q/k/v
  gemm_bf16_kernel<<<dim3(3072 / 128, M / 128), 256, 0, stream>>>(
      xb, wib, b_in, 1024, 3072, 0, qb, kb, vb, nullptr);

  // 2) attention: 4*16*64 = 4096 workgroups
  cron_attn_kernel<<<4 * 16 * 64, 128, 0, stream>>>(qb, kb, vb, attn);

  // 3) out projection: [16384,1024] x [1024,1024]^T -> f32 out
  gemm_bf16_kernel<<<dim3(1024 / 128, M / 128), 256, 0, stream>>>(
      attn, wob, b_out, 1024, 1024, 1, nullptr, nullptr, nullptr, out);
}
